// Localizer_50148038148776
// MI455X (gfx1250) — compile-verified, matrix operand order checked by reading, not run on hardware
//
#include <hip/hip_runtime.h>

// ---------------------------------------------------------------------------
// CDNA5 WMMA helpers (wave32, 16x16x32 f16 -> f32)
// ---------------------------------------------------------------------------
typedef __attribute__((ext_vector_type(16))) _Float16 v16h;
typedef __attribute__((ext_vector_type(2)))  __fp16   v2fp16;
typedef __attribute__((ext_vector_type(8)))  float    v8f;

union Frag16 { v16h v; unsigned u[8]; };
union FragF  { v8f  v; float   f[8]; };

__device__ __forceinline__ v8f wmma32(v16h a, v16h b, v8f c) {
    // (neg_a, A, neg_b, B, c_mod, C, reuse_a, reuse_b)
    return __builtin_amdgcn_wmma_f32_16x16x32_f16(false, a, false, b, (short)0, c,
                                                  false, false);
}

// Pack two f32 into one packed-f16 dword with a single v_cvt_pk instruction.
__device__ __forceinline__ unsigned pkhalf2(float a, float b) {
#if __has_builtin(__builtin_amdgcn_cvt_pkrtz)
    union { v2fp16 h; unsigned u; } p;
    p.h = __builtin_amdgcn_cvt_pkrtz(a, b);
    return p.u;
#else
    union { _Float16 h[2]; unsigned u; } p;
    p.h[0] = (_Float16)a; p.h[1] = (_Float16)b;
    return p.u;
#endif
}

// Low 32 bits of a generic pointer to LDS == wave-relative LDS byte offset.
__device__ __forceinline__ unsigned ldsoff(const void* p) {
    return (unsigned)(size_t)p;
}

// Async DMA: global -> LDS, 16 bytes per lane, tracked by ASYNCcnt.
__device__ __forceinline__ void async_ld_b128(unsigned lds, const void* g) {
    asm volatile("global_load_async_to_lds_b128 %0, %1, off"
                 :: "v"(lds), "v"((unsigned long long)(size_t)g) : "memory");
}
// Non-temporal flavor for stream-once data (img_feats: 400 MB > 192 MB L2).
__device__ __forceinline__ void async_ld_b128_nt(unsigned lds, const void* g) {
    asm volatile("global_load_async_to_lds_b128 %0, %1, off th:TH_LOAD_NT"
                 :: "v"(lds), "v"((unsigned long long)(size_t)g) : "memory");
}

// Fragment load from f16 LDS, row-major leading dim `ldk`.
// A: rc = row-tile base; B (stored col-major): rc = col-tile base.
// K-pair packing per ISA: lanes 0-15 pairs at k=0,2,4,6,16,18,20,22; lanes 16-31: +8.
__device__ __forceinline__ v16h ldfrag(const _Float16* base, int rc, int ldk, int k0) {
    const int lane = threadIdx.x & 31;
    const int m    = lane & 15;
    const int hi   = lane >> 4;
    const _Float16* p = base + (rc + m) * ldk + k0 + (hi ? 8 : 0);
    Frag16 f;
#pragma unroll
    for (int v = 0; v < 8; ++v) {
        const int kb = (v < 4) ? (2 * v) : (16 + 2 * (v - 4));
        f.u[v] = *(const unsigned*)(p + kb);
    }
    return f.v;
}

// A-fragment straight from an f32 LDS tile (filled by async DMA):
// ds_load_b64 pairs + v_cvt_pk at build time.
__device__ __forceinline__ v16h ldfragA_f32(const float* base, int rc, int ldk) {
    const int lane = threadIdx.x & 31;
    const int m    = lane & 15;
    const int hi   = lane >> 4;
    const float* p = base + (rc + m) * ldk + (hi ? 8 : 0);
    Frag16 f;
#pragma unroll
    for (int v = 0; v < 8; ++v) {
        const int kb = (v < 4) ? (2 * v) : (16 + 2 * (v - 4));
        const float2 t = *(const float2*)(p + kb);
        f.u[v] = pkhalf2(t.x, t.y);
    }
    return f.v;
}

// ---------------------------------------------------------------------------
// Device-scope software barrier (persistent cooperative LSTM kernel)
// ---------------------------------------------------------------------------
__device__ __forceinline__ void grid_barrier(unsigned* cnt, unsigned target) {
    __threadfence();      // release
    __syncthreads();
    if (threadIdx.x == 0) {
        atomicAdd(cnt, 1u);
        while (__hip_atomic_load(cnt, __ATOMIC_RELAXED, __HIP_MEMORY_SCOPE_AGENT) < target)
            __builtin_amdgcn_s_sleep(2);
    }
    __syncthreads();
    __threadfence();      // acquire
}

// ---------------------------------------------------------------------------
// Sizes
// ---------------------------------------------------------------------------
#define NB   64
#define LSEQ 512
#define KIMG 3200
#define FC   64
#define HID  256
#define NWG  8
#define KCH  100                    // 3200 / 32

#define LOGP_ELEMS (NB * LSEQ * 2)
#define FEAT_ELEMS (NB * LSEQ * HID)

// ---------------------------------------------------------------------------
// 1) phrase embed: emb = relu(phrase @ W_phr + b_phr)
// ---------------------------------------------------------------------------
__global__ __launch_bounds__(256) void emb_kernel(const float* __restrict__ phr,
                                                  const float* __restrict__ Wphr,
                                                  const float* __restrict__ bphr,
                                                  float* __restrict__ emb) {
    const int gid = blockIdx.x * 256 + threadIdx.x;   // 0..4095
    const int n = gid >> 6, f = gid & 63;
    float s = bphr[f];
    for (int d = 0; d < 300; ++d) s += phr[n * 300 + d] * Wphr[d * 64 + f];
    emb[gid] = s > 0.f ? s : 0.f;
}

// ---------------------------------------------------------------------------
// 2) fuse = relu(img @ W_cnn + b_cnn) * emb[n] -> f16
//    A tile: NT async-DMA double buffered f32 in LDS; B tile: packed f16.
// ---------------------------------------------------------------------------
__global__ __launch_bounds__(256) void cnn_fuse_kernel(const float* __restrict__ img,
                                                       const float* __restrict__ Wc,
                                                       const float* __restrict__ bc,
                                                       const float* __restrict__ emb,
                                                       _Float16* __restrict__ fuse16) {
    __shared__ alignas(16) float    sAf[2][128 * 32];  // 2 x 16 KB (async target)
    __shared__ alignas(16) _Float16 sB[64 * 32];       // col-major [col][k]
    __shared__ float sEmb[64];

    const int tid = threadIdx.x;
    const int r0  = blockIdx.x * 128;
    const int nb  = r0 >> 9;
    if (tid < 64) sEmb[tid] = emb[nb * 64 + tid];

    const int w = tid >> 5;

    // prologue: DMA chunk 0 into buffer 0 (4 async instructions per wave)
#pragma unroll
    for (int i = 0; i < 4; ++i) {
        const int idx = tid + i * 256;             // 0..1023 (16B segments)
        const int r = idx >> 3, seg = idx & 7;     // 8 segs x 16B per 32-float row
        async_ld_b128_nt(ldsoff(&sAf[0][idx * 4]),
                         img + (size_t)(r0 + r) * KIMG + seg * 4);
    }

    v8f acc[4] = {};
    for (int kc = 0; kc < KCH; ++kc) {
        const int k0 = kc * 32;
        __syncthreads();   // prior readers of next-buffer / sB are done
        if (kc + 1 < KCH) {
            const int k1 = k0 + 32;
#pragma unroll
            for (int i = 0; i < 4; ++i) {
                const int idx = tid + i * 256;
                const int r = idx >> 3, seg = idx & 7;
                async_ld_b128_nt(ldsoff(&sAf[(kc + 1) & 1][idx * 4]),
                                 img + (size_t)(r0 + r) * KIMG + k1 + seg * 4);
            }
        }
        // stage B chunk (32 x 64), packed pairs along k (source is k-strided)
#pragma unroll
        for (int i = 0; i < 4; ++i) {
            const int idx = tid + i * 256;         // 0..1023 pairs
            const int c = idx & 63, kp = idx >> 6; // kp = 0..15 -> k = 2*kp
            const unsigned pv = pkhalf2(Wc[(size_t)(k0 + 2 * kp) * FC + c],
                                        Wc[(size_t)(k0 + 2 * kp + 1) * FC + c]);
            *(unsigned*)&sB[c * 32 + 2 * kp] = pv;
        }
        if (kc + 1 < KCH) asm volatile("s_wait_asynccnt 4" ::: "memory");
        else              asm volatile("s_wait_asynccnt 0" ::: "memory");
        __syncthreads();

        const v16h a = ldfragA_f32(sAf[kc & 1], w * 16, 32);
#pragma unroll
        for (int nt = 0; nt < 4; ++nt) {
            const v16h b = ldfrag(sB, nt * 16, 32, 0);
            acc[nt] = wmma32(a, b, acc[nt]);
        }
    }

    const int lane = tid & 31, nIdx = lane & 15, hi = lane >> 4;
#pragma unroll
    for (int nt = 0; nt < 4; ++nt) {
        FragF d; d.v = acc[nt];
#pragma unroll
        for (int v = 0; v < 8; ++v) {
            const int row = r0 + w * 16 + v + (hi ? 8 : 0);
            const int col = nt * 16 + nIdx;
            float val = d.f[v] + bc[col];
            val = val > 0.f ? val : 0.f;
            val *= sEmb[col];
            fuse16[(size_t)row * FC + col] = (_Float16)val;
        }
    }
}

// ---------------------------------------------------------------------------
// 3) Persistent cooperative LSTM (8 WGs, 32 hidden units each, f16 weights in
//    LDS). x_t / h_t staged via async DMA; one device barrier per timestep.
// ---------------------------------------------------------------------------
__global__ __launch_bounds__(256) void lstm_kernel(const _Float16* __restrict__ fuse16,
                                                   const float* __restrict__ Wih,
                                                   const float* __restrict__ bih,
                                                   const float* __restrict__ Whh,
                                                   const float* __restrict__ bhh,
                                                   float* __restrict__ outFeats,
                                                   _Float16* __restrict__ h16,
                                                   unsigned* __restrict__ barrier) {
    __shared__ alignas(16) _Float16 sWih[128 * 64];    // 16 KB [lc][d]
    __shared__ alignas(16) _Float16 sWhh[128 * 256];   // 64 KB [lc][k]
    __shared__ alignas(16) _Float16 sX[64 * 64];       //  8 KB (async target)
    __shared__ alignas(16) _Float16 sH[64 * 256];      // 32 KB (async target)
    __shared__ alignas(16) float    sGates[64 * 128];  // 32 KB
    __shared__ float sBias[128];

    const int tid = threadIdx.x;
    const int j   = blockIdx.x;
    const int w   = tid >> 5;

    for (int idx = tid; idx < 128 * 64; idx += 256) {
        const int lc = idx >> 6, d = idx & 63;
        const int gcol = (lc >> 5) * 256 + j * 32 + (lc & 31);
        sWih[lc * 64 + d] = (_Float16)Wih[d * 1024 + gcol];
    }
    for (int idx = tid; idx < 128 * 256; idx += 256) {
        const int lc = idx >> 8, k = idx & 255;
        const int gcol = (lc >> 5) * 256 + j * 32 + (lc & 31);
        sWhh[lc * 256 + k] = (_Float16)Whh[k * 1024 + gcol];
    }
    if (tid < 128) {
        const int gcol = (tid >> 5) * 256 + j * 32 + (tid & 31);
        sBias[tid] = bih[gcol] + bhh[gcol];
    }
    for (int idx = tid; idx < 64 * 32; idx += 256) {   // h_0 = 0 (my slice, buf 0)
        const int r = idx >> 5, u = idx & 31;
        h16[r * 256 + j * 32 + u] = (_Float16)0.f;
    }

    float creg[8];
#pragma unroll
    for (int i = 0; i < 8; ++i) creg[i] = 0.f;

    unsigned phase = 0;
    grid_barrier(barrier, ++phase * NWG);

    const int mt   = w & 3;
    const int ntb  = (w >> 2) * 4;
    const int lane = tid & 31, nIdx = lane & 15, hi = lane >> 4;

    for (int t = 0; t < LSEQ; ++t) {
        const int cur = t & 1, nxt = cur ^ 1;

        // x_t: 64 rows x 128 B via async DMA (2 instr/wave)
#pragma unroll
        for (int i = 0; i < 2; ++i) {
            const int idx = tid + i * 256;             // 0..511
            const int n = idx >> 3, seg = idx & 7;
            async_ld_b128(ldsoff(sX + idx * 8),
                          fuse16 + ((size_t)n * LSEQ + t) * FC + seg * 8);
        }
        // h_t: 32 KB contiguous via async DMA (8 instr/wave)
        const _Float16* hsrc = h16 + (size_t)cur * (64 * 256);
#pragma unroll
        for (int i = 0; i < 8; ++i) {
            const int idx = tid + i * 256;             // 0..2047
            async_ld_b128(ldsoff(sH + idx * 8), hsrc + idx * 8);
        }
        asm volatile("s_wait_asynccnt 0" ::: "memory");
        __syncthreads();

        v8f acc[4] = {};
#pragma unroll
        for (int kc = 0; kc < 2; ++kc) {               // x @ W_ih (K=64)
            const v16h a = ldfrag(sX, mt * 16, 64, kc * 32);
#pragma unroll
            for (int q = 0; q < 4; ++q) {
                const v16h b = ldfrag(sWih, (ntb + q) * 16, 64, kc * 32);
                acc[q] = wmma32(a, b, acc[q]);
            }
        }
#pragma unroll 2
        for (int kc = 0; kc < 8; ++kc) {               // h @ W_hh (K=256)
            const v16h a = ldfrag(sH, mt * 16, 256, kc * 32);
#pragma unroll
            for (int q = 0; q < 4; ++q) {
                const v16h b = ldfrag(sWhh, (ntb + q) * 16, 256, kc * 32);
                acc[q] = wmma32(a, b, acc[q]);
            }
        }
#pragma unroll
        for (int q = 0; q < 4; ++q) {
            FragF d; d.v = acc[q];
#pragma unroll
            for (int v = 0; v < 8; ++v) {
                const int m  = mt * 16 + v + (hi ? 8 : 0);
                const int lc = (ntb + q) * 16 + nIdx;
                sGates[m * 128 + lc] = d.f[v] + sBias[lc];
            }
        }
        __syncthreads();

        // cell update: 8 consecutive units per thread -> vector stores
        _Float16* hdst = h16 + (size_t)nxt * (64 * 256);
        const int e0 = tid * 8;
        const int n = e0 >> 5, u0 = e0 & 31;
        float hv[8];
#pragma unroll
        for (int i = 0; i < 8; ++i) {
            const int u = u0 + i;
            float iv = sGates[n * 128 +      u];
            float fv = sGates[n * 128 + 32 + u];
            float gv = sGates[n * 128 + 64 + u];
            float ov = sGates[n * 128 + 96 + u];
            iv = 1.f / (1.f + expf(-iv));
            fv = 1.f / (1.f + expf(-fv));
            ov = 1.f / (1.f + expf(-ov));
            gv = tanhf(gv);
            const float c = fv * creg[i] + iv * gv;
            creg[i] = c;
            hv[i] = ov * tanhf(c);
        }
        float4* op = (float4*)(outFeats + ((size_t)n * LSEQ + t) * HID + j * 32 + u0);
        op[0] = make_float4(hv[0], hv[1], hv[2], hv[3]);
        op[1] = make_float4(hv[4], hv[5], hv[6], hv[7]);
        union { unsigned u[4]; uint4 q; } hp;
#pragma unroll
        for (int i = 0; i < 4; ++i) hp.u[i] = pkhalf2(hv[2 * i], hv[2 * i + 1]);
        *(uint4*)(hdst + n * 256 + j * 32 + u0) = hp.q;

        grid_barrier(barrier, ++phase * NWG);
    }
}

// ---------------------------------------------------------------------------
// 4) classifier + log_softmax(2). W_c1 LDS-resident f16 (packed staging).
// ---------------------------------------------------------------------------
__global__ __launch_bounds__(256) void classifier_kernel(const float* __restrict__ feats,
                                                         const float* __restrict__ Wc1,
                                                         const float* __restrict__ bc1,
                                                         const float* __restrict__ Wc2,
                                                         const float* __restrict__ bc2,
                                                         float* __restrict__ logp) {
    __shared__ alignas(16) _Float16 sA[128 * 32];
    __shared__ alignas(16) _Float16 sW[64 * 256];      // col-major [f][k]
    __shared__ alignas(16) float    sHid[128 * 64];

    const int tid = threadIdx.x;
    const int r0  = blockIdx.x * 128;
    const int w   = tid >> 5;

    for (int i = 0; i < 32; ++i) {                     // 8192 packed pairs
        const int idx = tid + i * 256;
        const int f = idx >> 7, kp = idx & 127;
        *(unsigned*)&sW[f * 256 + 2 * kp] =
            pkhalf2(Wc1[(size_t)(2 * kp) * 64 + f], Wc1[(size_t)(2 * kp + 1) * 64 + f]);
    }

    v8f acc[4] = {};
    for (int kc = 0; kc < 8; ++kc) {
        __syncthreads();
#pragma unroll
        for (int i = 0; i < 8; ++i) {                  // 2048 packed pairs (b64 loads)
            const int idx = tid + i * 256;
            const int r = idx >> 4, kp = idx & 15;
            const float2 f2 = *(const float2*)&feats[(size_t)(r0 + r) * HID + kc * 32 + 2 * kp];
            *(unsigned*)&sA[r * 32 + 2 * kp] = pkhalf2(f2.x, f2.y);
        }
        __syncthreads();
        const v16h a = ldfrag(sA, w * 16, 32, 0);
#pragma unroll
        for (int nt = 0; nt < 4; ++nt) {
            const v16h b = ldfrag(sW, nt * 16, 256, kc * 32);
            acc[nt] = wmma32(a, b, acc[nt]);
        }
    }

    const int lane = tid & 31, nIdx = lane & 15, hi = lane >> 4;
#pragma unroll
    for (int nt = 0; nt < 4; ++nt) {
        FragF d; d.v = acc[nt];
#pragma unroll
        for (int v = 0; v < 8; ++v) {
            const int m   = w * 16 + v + (hi ? 8 : 0);
            const int col = nt * 16 + nIdx;
            const float val = d.f[v] + bc1[col];
            sHid[m * 64 + col] = val > 0.f ? val : 0.f;
        }
    }
    __syncthreads();

    if (tid < 128) {
        const int row = r0 + tid;
        float z0 = bc2[0], z1 = bc2[1];
        for (int f = 0; f < 64; ++f) {
            const float hvv = sHid[tid * 64 + f];
            z0 += hvv * Wc2[f * 2 + 0];
            z1 += hvv * Wc2[f * 2 + 1];
        }
        const float mx  = fmaxf(z0, z1);
        const float lse = mx + logf(expf(z0 - mx) + expf(z1 - mx));
        logp[(size_t)row * 2 + 0] = z0 - lse;
        logp[(size_t)row * 2 + 1] = z1 - lse;
    }
}

// ---------------------------------------------------------------------------
// 5) selected[n] = output_feats[n, select_ixs[n], :]
// ---------------------------------------------------------------------------
__global__ __launch_bounds__(256) void select_kernel(const float* __restrict__ outFeats,
                                                     const int* __restrict__ ix,
                                                     float* __restrict__ sel) {
    const int n = blockIdx.x, h = threadIdx.x;
    const int t = ix[n];
    sel[n * HID + h] = outFeats[((size_t)n * LSEQ + t) * HID + h];
}

// ---------------------------------------------------------------------------
// Launcher
// ---------------------------------------------------------------------------
extern "C" void kernel_launch(void* const* d_in, const int* in_sizes, int n_in,
                              void* d_out, int out_size, void* d_ws, size_t ws_size,
                              hipStream_t stream) {
    const float* img   = (const float*)d_in[0];
    const float* phr   = (const float*)d_in[1];
    /* d_in[2] masks: unused, as in reference */
    const int*   selix = (const int*)d_in[3];
    const float* Wcnn  = (const float*)d_in[4];
    const float* bcnn  = (const float*)d_in[5];
    const float* Wphr  = (const float*)d_in[6];
    const float* bphr  = (const float*)d_in[7];
    const float* Wih   = (const float*)d_in[8];
    const float* bih   = (const float*)d_in[9];
    const float* Whh   = (const float*)d_in[10];
    const float* bhh   = (const float*)d_in[11];
    const float* Wc1   = (const float*)d_in[12];
    const float* bc1   = (const float*)d_in[13];
    const float* Wc2   = (const float*)d_in[14];
    const float* bc2   = (const float*)d_in[15];

    float* out      = (float*)d_out;
    float* logp     = out;
    float* outFeats = out + LOGP_ELEMS;
    float* selected = out + LOGP_ELEMS + FEAT_ELEMS;

    char* ws = (char*)d_ws;
    unsigned*  barrier = (unsigned*)ws;                         // 256 B
    float*     emb     = (float*)(ws + 256);                    // 16 KB
    _Float16*  fuse16  = (_Float16*)(ws + 32768);               // 4 MB
    _Float16*  h16     = (_Float16*)(ws + 32768 + 4194304);     // 64 KB (x2)

    (void)hipMemsetAsync(barrier, 0, sizeof(unsigned), stream);

    emb_kernel<<<16, 256, 0, stream>>>(phr, Wphr, bphr, emb);
    cnn_fuse_kernel<<<256, 256, 0, stream>>>(img, Wcnn, bcnn, emb, fuse16);
    lstm_kernel<<<NWG, 256, 0, stream>>>(fuse16, Wih, bih, Whh, bhh,
                                         outFeats, h16, barrier);
    classifier_kernel<<<256, 256, 0, stream>>>(outFeats, Wc1, bc1, Wc2, bc2, logp);
    select_kernel<<<NB, 256, 0, stream>>>(outFeats, selix, selected);
}
